// DifferentialMultiHeadSelfAttention_73564199846400
// MI455X (gfx1250) — matrix operand order repaired and checked by reading, not compiled, not run on hardware
//
#include <hip/hip_runtime.h>
#include <hip/hip_bf16.h>
#include <stdint.h>

// ---------------------------------------------------------------------------
// Differential multi-head self-attention for MI455X (gfx1250), wave32 + WMMA.
// B=1, S=2048, E=1024, H=16, DH=64, D=1024. All GEMMs via
// v_wmma_f32_16x16x32_bf16 (bf16 inputs, f32 accumulate). GEMM tiles are
// streamed into LDS with GLOBAL_LOAD_ASYNC_TO_LDS_B128 (ASYNCcnt),
// double-buffered.
// ---------------------------------------------------------------------------

#define S_LEN 2048
#define E_DIM 1024
#define H_NUM 16
#define DH_DIM 64
#define D_DIM 1024

typedef __attribute__((ext_vector_type(16))) __bf16 v16bf;
typedef __attribute__((ext_vector_type(8)))  float  v8f;

union ABReg { unsigned int u[8]; v16bf v; };

// A-matrix (16x32, 16-bit) K-pair base for VGPR v, lane half (ISA 7.12.2)
__device__ __forceinline__ int kpairA(int v, int half) {
  return ((v < 4) ? 0 : 16) + half * 8 + 2 * (v & 3);
}
// B-matrix (32x16, 16-bit) K-pair base for VGPR v, lane half (same k order as A)
__device__ __forceinline__ int kpairB(int v, int half) {
  return ((v < 4) ? 0 : 16) + half * 8 + 2 * (v & 3);
}

__device__ __forceinline__ unsigned short f2bf(float f) {
  unsigned int u = __float_as_uint(f);
  u += 0x7fffu + ((u >> 16) & 1u);   // round-to-nearest-even
  return (unsigned short)(u >> 16);
}
__device__ __forceinline__ unsigned int ldu32(const unsigned short* p) {
  return *(const unsigned int*)p;
}

// --- CDNA5 async global->LDS copy (16B per lane), tracked by ASYNCcnt ------
__device__ __forceinline__ void async_copy16(unsigned lds_off, const void* g) {
  unsigned long long ga = (unsigned long long)(uintptr_t)g;
  asm volatile("global_load_async_to_lds_b128 %0, %1, off"
               :: "v"(lds_off), "v"(ga)
               : "memory");
}
__device__ __forceinline__ void wait_async0() {
  asm volatile("s_wait_asynccnt 0x0" ::: "memory");
}

// ---------------------------------------------------------------------------
// Kernel 0: f32 -> bf16 convert
// ---------------------------------------------------------------------------
__global__ void cvt_f32_to_bf16(const float* __restrict__ src,
                                unsigned short* __restrict__ dst, int n) {
  int i = blockIdx.x * blockDim.x + threadIdx.x;
  if (i < n) dst[i] = f2bf(src[i]);
}

// ---------------------------------------------------------------------------
// Staged 64x64 GEMM core: C[m0:m0+64, n0:n0+64] = A[M,K] . B[N,K]^T.
// Block = 128 threads (4 waves); wave w owns rows m0+w*16..+16. A/B k-slices
// (64x32 bf16, 4KB each) are async-staged into LDS, ping-pong buffered.
// acc[4] per wave: four 16x16 f32 tiles across the 64 columns.
// ---------------------------------------------------------------------------
__device__ __forceinline__ void stage_tiles(
    const unsigned short* __restrict__ Ag,
    const unsigned short* __restrict__ Bg,
    int m0, int n0, int K, int kc, int tid,
    unsigned short (*aT)[32], unsigned short (*bT)[32]) {   // each [64][32]
#pragma unroll
  for (int i = 0; i < 2; ++i) {
    const int c   = tid + i * 128;      // 16B chunk id, 256 chunks per tile
    const int row = c >> 2;             // 4 chunks (64B) per row
    const int ko  = (c & 3) * 8;        // bf16 element offset in k
    async_copy16((unsigned)(uintptr_t)&aT[row][ko],
                 Ag + (size_t)(m0 + row) * K + kc + ko);
    async_copy16((unsigned)(uintptr_t)&bT[row][ko],
                 Bg + (size_t)(n0 + row) * K + kc + ko);
  }
}

__device__ __forceinline__ void gemm_core_64x64(
    const unsigned short* __restrict__ Ag,
    const unsigned short* __restrict__ Bg,
    int m0, int n0, int K,
    unsigned short (*aT)[64][32],       // [2][64][32] LDS
    unsigned short (*bT)[64][32],       // [2][64][32] LDS
    v8f* acc) {                         // [4]
  const int tid  = threadIdx.x;
  const int w    = tid >> 5;
  const int lane = tid & 31;
  const int half = lane >> 4;
  const int lm   = lane & 15;
  const int nIter = K / 32;

  stage_tiles(Ag, Bg, m0, n0, K, 0, tid, aT[0], bT[0]);

  for (int it = 0; it < nIter; ++it) {
    wait_async0();        // this wave's in-flight batch (tiles for `it`) done
    __syncthreads();      // all waves done waiting AND done reading buf[it^1]
    if (it + 1 < nIter)
      stage_tiles(Ag, Bg, m0, n0, K, (it + 1) * 32, tid,
                  aT[(it + 1) & 1], bT[(it + 1) & 1]);
    const int buf = it & 1;

    ABReg a;
#pragma unroll
    for (int v = 0; v < 8; ++v)
      a.u[v] = ldu32(&aT[buf][w * 16 + lm][kpairA(v, half)]);
#pragma unroll
    for (int nt = 0; nt < 4; ++nt) {
      ABReg bm;
#pragma unroll
      for (int v = 0; v < 8; ++v)
        bm.u[v] = ldu32(&bT[buf][nt * 16 + lm][kpairB(v, half)]);
      acc[nt] = __builtin_amdgcn_wmma_f32_16x16x32_bf16(
          false, a.v, false, bm.v, (short)0, acc[nt], false, false);
    }
  }
}

// ---------------------------------------------------------------------------
// Kernel 1: fused projection GEMM.
// out[s, col] = sum_e x[s,e] * W[col, e] + b[col],  col = p*1024 + h*64 + d
// (wbf is [5*16*64, 1024] row-major: row index == col). Results stored bf16
// as projbf[((p*16+h)*2048 + s)*64 + d]. Grid (32, 80), block 128.
// ---------------------------------------------------------------------------
__global__ void __launch_bounds__(128)
proj_gemm(const unsigned short* __restrict__ xbf,
          const unsigned short* __restrict__ wbf,
          const float* __restrict__ b0, const float* __restrict__ b1,
          const float* __restrict__ b2, const float* __restrict__ b3,
          const float* __restrict__ b4,
          unsigned short* __restrict__ projbf) {
  __shared__ __align__(16) unsigned short aT[2][64][32];
  __shared__ __align__(16) unsigned short bT[2][64][32];
  const int tid  = threadIdx.x;
  const int w    = tid >> 5;
  const int lane = tid & 31;
  const int half = lane >> 4;
  const int lm   = lane & 15;
  const int m0 = blockIdx.x * 64;
  const int n0 = blockIdx.y * 64;

  v8f acc[4] = {};
  gemm_core_64x64(xbf, wbf, m0, n0, E_DIM, aT, bT, acc);

#pragma unroll
  for (int nt = 0; nt < 4; ++nt) {
    const int col = n0 + nt * 16 + lm;
    const int p   = col >> 10;
    const int rem = col & 1023;
    const int dd  = rem & 63;
    const float* bp = (p == 0) ? b0 : (p == 1) ? b1 : (p == 2) ? b2
                     : (p == 3) ? b3 : b4;
    const float bias = bp[rem];                      // b[h*64+d]
    const size_t hbase = (size_t)(col - dd) * S_LEN; // (p*16+h)*64*2048
#pragma unroll
    for (int r = 0; r < 8; ++r) {
      const int s = m0 + w * 16 + r + half * 8;
      projbf[hbase + (size_t)s * DH_DIM + dd] = f2bf(acc[nt][r] + bias);
    }
  }
}

// ---------------------------------------------------------------------------
// Kernel 2: differential causal flash attention (two softmax streams).
// Block = 128 threads (4 waves), grid (S/64, H). Wave owns 16 query rows.
// ---------------------------------------------------------------------------
__device__ __forceinline__ void attn_stream_step(
    const ABReg* aQ,                       // [2] A-regs (DH halves)
    const unsigned short* __restrict__ kb, // K proj [S,64] bf16
    const unsigned short* __restrict__ vb, // V proj [S,64] bf16
    int t0, int q0, int half, int lm,
    unsigned short (*plds)[32],            // per-wave [16][32] staging
    float* mrow, float* lrow, v8f* o) {    // o[4]
  // scores: 16 q x 32 keys, two 16x16 subtiles, K-reduce over DH=64
  v8f sc[2];
#pragma unroll
  for (int n0t = 0; n0t < 2; ++n0t) {
    v8f c = {};
#pragma unroll
    for (int kh = 0; kh < 2; ++kh) {
      ABReg bm;
#pragma unroll
      for (int v = 0; v < 8; ++v) {
        const int dh  = kh * 32 + kpairB(v, half);
        const int key = t0 + n0t * 16 + lm;
        bm.u[v] = ldu32(kb + (size_t)key * DH_DIM + dh);
      }
      c = __builtin_amdgcn_wmma_f32_16x16x32_bf16(
          false, aQ[kh].v, false, bm.v, (short)0, c, false, false);
    }
    sc[n0t] = c;
  }
  // scale + causal mask
#pragma unroll
  for (int n0t = 0; n0t < 2; ++n0t)
#pragma unroll
    for (int r = 0; r < 8; ++r) {
      const int q   = q0 + r + half * 8;
      const int key = t0 + n0t * 16 + lm;
      const float v = sc[n0t][r] * 0.125f;  // 1/sqrt(64)
      sc[n0t][r] = (key > q) ? -1.0e30f : v;
    }
  // running max / rescale factor per row (row lives in one 16-lane half)
  float alpha[8];
#pragma unroll
  for (int r = 0; r < 8; ++r) {
    float mx = fmaxf(sc[0][r], sc[1][r]);
#pragma unroll
    for (int off = 8; off >= 1; off >>= 1)
      mx = fmaxf(mx, __shfl_xor(mx, off, 16));
    const float mn = fmaxf(mrow[r], mx);
    alpha[r] = __expf(mrow[r] - mn);
    mrow[r]  = mn;
  }
  // P = exp(s - m), row sums
#pragma unroll
  for (int r = 0; r < 8; ++r) {
    const float p0 = __expf(sc[0][r] - mrow[r]);
    const float p1 = __expf(sc[1][r] - mrow[r]);
    sc[0][r] = p0; sc[1][r] = p1;
    float rs = p0 + p1;
#pragma unroll
    for (int off = 8; off >= 1; off >>= 1)
      rs += __shfl_xor(rs, off, 16);
    lrow[r] = lrow[r] * alpha[r] + rs;
  }
  // C/D layout -> A layout via wave-local LDS staging (bf16)
#pragma unroll
  for (int n0t = 0; n0t < 2; ++n0t)
#pragma unroll
    for (int r = 0; r < 8; ++r)
      plds[r + half * 8][n0t * 16 + lm] = f2bf(sc[n0t][r]);
  ABReg pa;
#pragma unroll
  for (int v = 0; v < 8; ++v)
    pa.u[v] = ldu32(&plds[lm][kpairA(v, half)]);
  // O = O*alpha + P @ V  (4 dh tiles of 16)
#pragma unroll
  for (int nt = 0; nt < 4; ++nt) {
#pragma unroll
    for (int r = 0; r < 8; ++r) o[nt][r] *= alpha[r];
    ABReg bv;
#pragma unroll
    for (int v = 0; v < 8; ++v) {
      const int kk = kpairB(v, half);    // key pair along K dim
      const unsigned int lo = vb[(size_t)(t0 + kk)     * DH_DIM + nt * 16 + lm];
      const unsigned int hi = vb[(size_t)(t0 + kk + 1) * DH_DIM + nt * 16 + lm];
      bv.u[v] = lo | (hi << 16);
    }
    o[nt] = __builtin_amdgcn_wmma_f32_16x16x32_bf16(
        false, pa.v, false, bv.v, (short)0, o[nt], false, false);
  }
}

__global__ void __launch_bounds__(128)
diff_attn(const unsigned short* __restrict__ projbf,
          const float* __restrict__ lam,
          float* __restrict__ cat) {       // [S, D] f32
  __shared__ __align__(8) unsigned short plds[4][16][32];
  const int tid  = threadIdx.x;
  const int w    = tid >> 5;
  const int lane = tid & 31;
  const int half = lane >> 4;
  const int lm   = lane & 15;
  const int h  = blockIdx.y;
  const int q0 = blockIdx.x * 64 + w * 16;
  const float lamh = lam[h];

  const size_t hs = (size_t)S_LEN * DH_DIM;
  const unsigned short* qb1 = projbf + (size_t)(0 * H_NUM + h) * hs;
  const unsigned short* qb2 = projbf + (size_t)(1 * H_NUM + h) * hs;
  const unsigned short* kb1 = projbf + (size_t)(2 * H_NUM + h) * hs;
  const unsigned short* kb2 = projbf + (size_t)(3 * H_NUM + h) * hs;
  const unsigned short* vb  = projbf + (size_t)(4 * H_NUM + h) * hs;

  // load Q1/Q2 in A-layout, both DH halves
  ABReg a1[2], a2[2];
#pragma unroll
  for (int kh = 0; kh < 2; ++kh)
#pragma unroll
    for (int v = 0; v < 8; ++v) {
      const int k = kh * 32 + kpairA(v, half);
      a1[kh].u[v] = ldu32(qb1 + (size_t)(q0 + lm) * DH_DIM + k);
      a2[kh].u[v] = ldu32(qb2 + (size_t)(q0 + lm) * DH_DIM + k);
    }

  v8f o1[4] = {}, o2[4] = {};
  float m1[8], m2[8], l1[8], l2[8];
#pragma unroll
  for (int r = 0; r < 8; ++r) {
    m1[r] = -1.0e30f; m2[r] = -1.0e30f; l1[r] = 0.f; l2[r] = 0.f;
  }

  const int qmax = q0 + 15;
  for (int t0 = 0; t0 <= qmax; t0 += 32) {
    attn_stream_step(a1, kb1, vb, t0, q0, half, lm, plds[w], m1, l1, o1);
    attn_stream_step(a2, kb2, vb, t0, q0, half, lm, plds[w], m2, l2, o2);
  }

#pragma unroll
  for (int nt = 0; nt < 4; ++nt)
#pragma unroll
    for (int r = 0; r < 8; ++r) {
      const int s  = q0 + r + half * 8;
      const int dh = nt * 16 + lm;
      const float val = o1[nt][r] / l1[r] - lamh * (o2[nt][r] / l2[r]);
      cat[(size_t)s * D_DIM + h * DH_DIM + dh] = val;
    }
}

// ---------------------------------------------------------------------------
// Kernel 3: GroupNorm (num_groups = H): normalize each head's [S, 64] slab.
// One block per head, 256 threads. Writes bf16 xn (gn affine applied).
// ---------------------------------------------------------------------------
__global__ void __launch_bounds__(256)
group_norm(const float* __restrict__ cat,
           const float* __restrict__ gn_w, const float* __restrict__ gn_b,
           unsigned short* __restrict__ xnbf) {
  __shared__ float ssum[256], ssq[256];
  const int h = blockIdx.x, tid = threadIdx.x;
  const int N = S_LEN * DH_DIM;
  float sum = 0.f, sq = 0.f;
  for (int i = tid; i < N; i += 256) {
    const int s = i >> 6, d = i & 63;
    const float v = cat[(size_t)s * D_DIM + h * DH_DIM + d];
    sum += v; sq += v * v;
  }
  ssum[tid] = sum; ssq[tid] = sq;
  __syncthreads();
  for (int st = 128; st > 0; st >>= 1) {
    if (tid < st) { ssum[tid] += ssum[tid + st]; ssq[tid] += ssq[tid + st]; }
    __syncthreads();
  }
  const float mean = ssum[0] / (float)N;
  const float var  = ssq[0] / (float)N - mean * mean;
  const float inv  = rsqrtf(var + 1e-5f);
  for (int i = tid; i < N; i += 256) {
    const int s = i >> 6, d = i & 63;
    const int col = h * DH_DIM + d;
    const float v = (cat[(size_t)s * D_DIM + col] - mean) * inv * gn_w[col]
                    + gn_b[col];
    xnbf[(size_t)s * D_DIM + col] = f2bf(v);
  }
}

// ---------------------------------------------------------------------------
// Kernel 4: output projection out[s,e] = xn[s,:] . Wo[e,:] + bo[e], f32 out.
// Same staged tiling as proj_gemm; grid (32, 16), block 128.
// ---------------------------------------------------------------------------
__global__ void __launch_bounds__(128)
out_gemm(const unsigned short* __restrict__ xnbf,
         const unsigned short* __restrict__ wobf,
         const float* __restrict__ bo, float* __restrict__ out) {
  __shared__ __align__(16) unsigned short aT[2][64][32];
  __shared__ __align__(16) unsigned short bT[2][64][32];
  const int tid  = threadIdx.x;
  const int w    = tid >> 5;
  const int lane = tid & 31;
  const int half = lane >> 4;
  const int lm   = lane & 15;
  const int m0 = blockIdx.x * 64;
  const int n0 = blockIdx.y * 64;

  v8f acc[4] = {};
  gemm_core_64x64(xnbf, wobf, m0, n0, D_DIM, aT, bT, acc);

#pragma unroll
  for (int nt = 0; nt < 4; ++nt) {
    const int col = n0 + nt * 16 + lm;
    const float bias = bo[col];
#pragma unroll
    for (int r = 0; r < 8; ++r) {
      const int s = m0 + w * 16 + r + half * 8;
      out[(size_t)s * E_DIM + col] = acc[nt][r] + bias;
    }
  }
}

// ---------------------------------------------------------------------------
// Host launcher
// ---------------------------------------------------------------------------
extern "C" void kernel_launch(void* const* d_in, const int* in_sizes, int n_in,
                              void* d_out, int out_size, void* d_ws,
                              size_t ws_size, hipStream_t stream) {
  (void)in_sizes; (void)n_in; (void)out_size; (void)ws_size;

  const float* x   = (const float*)d_in[0];
  // d_in[1] = causal mask (recomputed analytically on device)
  const float* Wq1 = (const float*)d_in[2];
  const float* bq1 = (const float*)d_in[3];
  const float* Wq2 = (const float*)d_in[4];
  const float* bq2 = (const float*)d_in[5];
  const float* Wk1 = (const float*)d_in[6];
  const float* bk1 = (const float*)d_in[7];
  const float* Wk2 = (const float*)d_in[8];
  const float* bk2 = (const float*)d_in[9];
  const float* Wv  = (const float*)d_in[10];
  const float* bv  = (const float*)d_in[11];
  const float* lam = (const float*)d_in[12];
  const float* gnw = (const float*)d_in[13];
  const float* gnb = (const float*)d_in[14];
  const float* Wo  = (const float*)d_in[15];
  const float* bo  = (const float*)d_in[16];
  float* out = (float*)d_out;

  // Workspace layout (element counts)
  const size_t nX  = (size_t)S_LEN * E_DIM;                 // 2,097,152
  const size_t nW1 = (size_t)H_NUM * DH_DIM * E_DIM;        // 1,048,576
  const size_t nW  = 5 * nW1;                               // 5,242,880
  const size_t nWo = (size_t)E_DIM * D_DIM;                 // 1,048,576
  const size_t nP  = 5 * (size_t)H_NUM * S_LEN * DH_DIM;    // 10,485,760
  const size_t nXn = (size_t)S_LEN * D_DIM;                 // 2,097,152

  unsigned short* xbf    = (unsigned short*)d_ws;
  unsigned short* wbf    = xbf + nX;
  unsigned short* wobf   = wbf + nW;
  unsigned short* projbf = wobf + nWo;
  unsigned short* xnbf   = projbf + nP;
  float*          cat    = (float*)(xnbf + nXn);            // [S, D] f32

  // 0) f32 -> bf16 conversions
  cvt_f32_to_bf16<<<(int)((nX + 255) / 256), 256, 0, stream>>>(x, xbf, (int)nX);
  cvt_f32_to_bf16<<<(int)((nW1 + 255) / 256), 256, 0, stream>>>(Wq1, wbf + 0 * nW1, (int)nW1);
  cvt_f32_to_bf16<<<(int)((nW1 + 255) / 256), 256, 0, stream>>>(Wq2, wbf + 1 * nW1, (int)nW1);
  cvt_f32_to_bf16<<<(int)((nW1 + 255) / 256), 256, 0, stream>>>(Wk1, wbf + 2 * nW1, (int)nW1);
  cvt_f32_to_bf16<<<(int)((nW1 + 255) / 256), 256, 0, stream>>>(Wk2, wbf + 3 * nW1, (int)nW1);
  cvt_f32_to_bf16<<<(int)((nW1 + 255) / 256), 256, 0, stream>>>(Wv,  wbf + 4 * nW1, (int)nW1);
  cvt_f32_to_bf16<<<(int)((nWo + 255) / 256), 256, 0, stream>>>(Wo, wobf, (int)nWo);

  // 1) fused q1/q2/k1/k2/v projection GEMM (WMMA bf16, async-LDS staged)
  proj_gemm<<<dim3(S_LEN / 64, (5 * D_DIM) / 64), 128, 0, stream>>>(
      xbf, wbf, bq1, bq2, bk1, bk2, bv, projbf);

  // 2) differential causal flash attention (WMMA bf16, f32 softmax state)
  diff_attn<<<dim3(S_LEN / 64, H_NUM), 128, 0, stream>>>(projbf, lam, cat);

  // 3) GroupNorm over each head's (DH, S) slab
  group_norm<<<H_NUM, 256, 0, stream>>>(cat, gnw, gnb, xnbf);

  // 4) output projection (WMMA bf16, async-LDS staged) -> f32 d_out
  out_gemm<<<dim3(S_LEN / 64, D_DIM / 64), 128, 0, stream>>>(xnbf, wobf, bo, out);
}
